// Net_79602923864324
// MI455X (gfx1250) — matrix operand order repaired and checked by reading, not compiled, not run on hardware
//
#include <hip/hip_runtime.h>
#include <hip/hip_bf16.h>

typedef __attribute__((ext_vector_type(16))) _Float16 v16h;
typedef __attribute__((ext_vector_type(8)))  _Float16 v8h;
typedef __attribute__((ext_vector_type(8)))  float    v8f;

#define NT     256
#define WAVES  8

// tanh via fast exp (CDNA5 trans units); inputs are O(1) so clamp is safe
__device__ __forceinline__ float fast_tanh(float x) {
  float xx = fminf(fmaxf(x, -15.0f), 15.0f);
  float e2 = __expf(2.0f * xx);
  return (e2 - 1.0f) / (e2 + 1.0f);
}

// Fill LDS with a KxN (row-major) f32 weight matrix converted to f16 and
// pre-swizzled into WMMA B-fragment order:
//   fragment(fK,fN) = 512 halves laid out [lane(32)][half(16)]
//   n = 16*fN + (lane&15) ; k = 32*fK + ((lane>=16)?8:0) + (h<8 ? h : h+8)
__device__ __forceinline__ void fill_wfrag(_Float16* dst, const float* W,
                                           int K, int N, int nfN, int nfrag,
                                           int tid) {
  int total = nfrag * 512;
  for (int e = tid; e < total; e += NT) {
    int frag = e >> 9, slot = e & 511;
    int lane = slot >> 4, h = slot & 15;
    int fK = frag / nfN, fN = frag - fK * nfN;
    int n = 16 * fN + (lane & 15);
    int k = 32 * fK + ((lane >> 4) << 3) + (h < 8 ? h : h + 8);
    float v = (k < K && n < N) ? W[k * N + n] : 0.0f;
    dst[e] = (_Float16)v;
  }
}

// w_l2 is [N=50][H=100][A=2]; flattened GEMM B is [K=100 rows=h][cols c=2n+a]
__device__ __forceinline__ void fill_wfrag_l2(_Float16* dst, const float* W, int tid) {
  for (int e = tid; e < 28 * 512; e += NT) {
    int frag = e >> 9, slot = e & 511;
    int lane = slot >> 4, h = slot & 15;
    int fK = frag / 7, fN = frag - fK * 7;
    int c = 16 * fN + (lane & 15);
    int k = 32 * fK + ((lane >> 4) << 3) + (h < 8 ? h : h + 8);
    float v = (k < 100 && c < 100) ? W[(c >> 1) * 200 + k * 2 + (c & 1)] : 0.0f;
    dst[e] = (_Float16)v;
  }
}

__global__ __launch_bounds__(NT) void rbf_dqn_kernel(
    const float* __restrict__ s,   const float* __restrict__ a,
    const float* __restrict__ w_v1, const float* __restrict__ b_v1,
    const float* __restrict__ w_v2, const float* __restrict__ b_v2,
    const float* __restrict__ w_l1, const float* __restrict__ b_l1,
    const float* __restrict__ w_l2, const float* __restrict__ b_l2,
    float* __restrict__ out, int Bn)
{
  // Weights (B fragments, f16, pre-swizzled) + biases (f32)
  __shared__ __attribute__((aligned(32))) _Float16 sWv1[7  * 512];  //  7 KB
  __shared__ __attribute__((aligned(32))) _Float16 sWl1[7  * 512];  //  7 KB
  __shared__ __attribute__((aligned(32))) _Float16 sWv2[16 * 512];  // 16 KB
  __shared__ __attribute__((aligned(32))) _Float16 sWl2[28 * 512];  // 28 KB
  __shared__ float sBv1[112], sBl1[112], sBv2[64], sBl2[112];
  // Per-wave staging: hv/hl as f16 [16][128], reused as f32 logits [16][64]
  __shared__ __attribute__((aligned(32))) _Float16 sStage[WAVES][16 * 128]; // 32 KB

  const int tid = threadIdx.x;

  fill_wfrag(sWv1, w_v1, 10, 100, 7,  7, tid);
  fill_wfrag(sWl1, w_l1, 10, 100, 7,  7, tid);
  fill_wfrag(sWv2, w_v2, 100, 50, 4, 16, tid);
  fill_wfrag_l2(sWl2, w_l2, tid);
  for (int i = tid; i < 112; i += NT) sBv1[i] = (i < 100) ? b_v1[i] : 0.0f;
  for (int i = tid; i < 112; i += NT) sBl1[i] = (i < 100) ? b_l1[i] : 0.0f;
  for (int i = tid; i < 64;  i += NT) sBv2[i] = (i <  50) ? b_v2[i] : 0.0f;
  for (int i = tid; i < 112; i += NT) sBl2[i] = (i < 100) ? b_l2[i] : 0.0f;
  __syncthreads();

  const int lane = tid & 31;
  const int wave = tid >> 5;
  _Float16* stg = sStage[wave];

  const int row  = lane & 15;          // A-fragment row (M)
  const int kb   = (lane >> 4) * 8;    // A-fragment K base
  const int colb = lane & 15;          // C/D fragment column (N)
  const int mhb  = (lane >> 4) * 8;    // C/D fragment row base (M)

  const int ntiles = (Bn + 15) >> 4;
  const int tstep  = gridDim.x * WAVES;

  for (int tile = blockIdx.x * WAVES + wave; tile < ntiles; tile += tstep) {
    const long m0 = (long)tile * 16;

    // prefetch next tile's s rows (lowers to global_prefetch_b8)
    {
      long nm = (long)(tile + tstep);
      long off = nm * 160 + lane * 16;
      if (nm < ntiles && off + 16 <= (long)Bn * 10)
        __builtin_prefetch(s + off, 0, 0);
    }

    // ---- s A-fragment (16x32 f16, K=10 valid) from global ----
    v16h sa;
    {
      long r = m0 + row; if (r >= Bn) r = Bn - 1;
      const float* srow = s + r * 10;
      #pragma unroll
      for (int h = 0; h < 16; ++h) {
        int k = kb + (h < 8 ? h : h + 8);
        sa[h] = (_Float16)((k < 10) ? srow[k] : 0.0f);
      }
    }

    // ---- value tower L1: hv = relu(s@Wv1 + b) -> staged f16 [16][128] ----
    #pragma unroll
    for (int f = 0; f < 7; ++f) {
      float bb = sBv1[16 * f + colb];
      v8f c;
      #pragma unroll
      for (int i = 0; i < 8; ++i) c[i] = bb;
      v8f acc = __builtin_amdgcn_wmma_f32_16x16x32_f16(
          false, sa, false, *(const v16h*)(sWv1 + f * 512 + lane * 16),
          (short)0, c, false, false);
      int col = 16 * f + colb;
      #pragma unroll
      for (int v = 0; v < 8; ++v) {
        float x = fmaxf(acc[v], 0.0f);
        stg[(mhb + v) * 128 + col] = (col < 100) ? (_Float16)x : (_Float16)0.0f;
      }
    }
    #pragma unroll
    for (int v = 0; v < 8; ++v)                     // pad cols 112..127 once
      stg[(mhb + v) * 128 + 112 + colb] = (_Float16)0.0f;

    // ---- value tower L2: centroid_values = hv@Wv2 + b   [16 x 64(50)] ----
    v8f accv[4];
    #pragma unroll
    for (int f = 0; f < 4; ++f) {
      float bb = sBv2[16 * f + colb];
      #pragma unroll
      for (int i = 0; i < 8; ++i) accv[f][i] = bb;
    }
    #pragma unroll
    for (int kt = 0; kt < 4; ++kt) {
      const _Float16* p = stg + row * 128 + kt * 32 + kb;
      union { v16h v; v8h h[2]; } ua;
      ua.h[0] = *(const v8h*)p;          // K = base .. base+7
      ua.h[1] = *(const v8h*)(p + 16);   // K = base+16 .. base+23
      #pragma unroll
      for (int f = 0; f < 4; ++f)
        accv[f] = __builtin_amdgcn_wmma_f32_16x16x32_f16(
            false, ua.v, false,
            *(const v16h*)(sWv2 + (kt * 4 + f) * 512 + lane * 16),
            (short)0, accv[f], false, false);
    }

    // ---- location tower L1: hl = relu(s@Wl1 + b) -> restage ----
    #pragma unroll
    for (int f = 0; f < 7; ++f) {
      float bb = sBl1[16 * f + colb];
      v8f c;
      #pragma unroll
      for (int i = 0; i < 8; ++i) c[i] = bb;
      v8f acc = __builtin_amdgcn_wmma_f32_16x16x32_f16(
          false, sa, false, *(const v16h*)(sWl1 + f * 512 + lane * 16),
          (short)0, c, false, false);
      int col = 16 * f + colb;
      #pragma unroll
      for (int v = 0; v < 8; ++v) {
        float x = fmaxf(acc[v], 0.0f);
        stg[(mhb + v) * 128 + col] = (col < 100) ? (_Float16)x : (_Float16)0.0f;
      }
    }

    // ---- location tower L2: [16 x 112(100)] pre-activation ----
    v8f accl[7];
    #pragma unroll
    for (int f = 0; f < 7; ++f) {
      float bb = sBl2[16 * f + colb];
      #pragma unroll
      for (int i = 0; i < 8; ++i) accl[f][i] = bb;
    }
    #pragma unroll
    for (int kt = 0; kt < 4; ++kt) {
      const _Float16* p = stg + row * 128 + kt * 32 + kb;
      union { v16h v; v8h h[2]; } ua;
      ua.h[0] = *(const v8h*)p;
      ua.h[1] = *(const v8h*)(p + 16);
      #pragma unroll
      for (int f = 0; f < 7; ++f)
        accl[f] = __builtin_amdgcn_wmma_f32_16x16x32_f16(
            false, ua.v, false,
            *(const v16h*)(sWl2 + (kt * 7 + f) * 512 + lane * 16),
            (short)0, accl[f], false, false);
    }

    // ---- epilogue: loc = 5*tanh(.), logits = -3*||loc - a||  ----
    float av_[8];                       // a[row][axis], axis = lane&1
    #pragma unroll
    for (int v = 0; v < 8; ++v) {
      long r = m0 + mhb + v; if (r >= Bn) r = Bn - 1;
      av_[v] = a[r * 2 + (lane & 1)];
    }
    float* lst = (float*)stg;           // reuse stage as f32 logits [16][64]
    #pragma unroll
    for (int f = 0; f < 7; ++f) {
      int col = 16 * f + colb;          // col parity == lane parity
      #pragma unroll
      for (int v = 0; v < 8; ++v) {
        float t  = 5.0f * fast_tanh(accl[f][v]);
        float d  = t - av_[v];
        float sq = d * d;
        float s2 = sq + __shfl_xor(sq, 1, 32);   // pair (2n, 2n+1)
        if (((lane & 1) == 0) && col < 100)
          lst[(mhb + v) * 64 + (col >> 1)] = -3.0f * sqrtf(s2);
      }
    }

    // ---- softmax over n (0..49) per row, weighted sum with values ----
    #pragma unroll
    for (int v = 0; v < 8; ++v) {
      int r = mhb + v;
      float lg[4];
      float mx = -1e30f;
      #pragma unroll
      for (int fv = 0; fv < 4; ++fv) {
        int n = 16 * fv + colb;
        float L = (n < 50) ? lst[r * 64 + n] : -1e30f;
        lg[fv] = L;
        mx = fmaxf(mx, L);
      }
      #pragma unroll
      for (int off = 8; off > 0; off >>= 1)      // within 16-lane half group
        mx = fmaxf(mx, __shfl_xor(mx, off, 32));
      float se = 0.0f, sw = 0.0f;
      #pragma unroll
      for (int fv = 0; fv < 4; ++fv) {
        float e = __expf(lg[fv] - mx);           // invalid n -> exp(-huge)=0
        se += e;
        sw += e * accv[fv][v];
      }
      #pragma unroll
      for (int off = 8; off > 0; off >>= 1) {
        se += __shfl_xor(se, off, 32);
        sw += __shfl_xor(sw, off, 32);
      }
      if ((lane & 15) == 0) {
        long orow = m0 + r;
        if (orow < Bn) out[orow] = sw / se;
      }
    }
  }
}

extern "C" void kernel_launch(void* const* d_in, const int* in_sizes, int n_in,
                              void* d_out, int out_size, void* d_ws, size_t ws_size,
                              hipStream_t stream) {
  const float* s    = (const float*)d_in[0];
  const float* a_   = (const float*)d_in[1];
  const float* wv1  = (const float*)d_in[2];
  const float* bv1  = (const float*)d_in[3];
  const float* wv2  = (const float*)d_in[4];
  const float* bv2  = (const float*)d_in[5];
  const float* wl1  = (const float*)d_in[6];
  const float* bl1  = (const float*)d_in[7];
  const float* wl2  = (const float*)d_in[8];
  const float* bl2  = (const float*)d_in[9];
  float* out = (float*)d_out;
  (void)d_ws; (void)ws_size; (void)n_in; (void)out_size;

  int Bn = in_sizes[0] / 10;
  int ntiles = (Bn + 15) >> 4;
  int slots  = (ntiles + WAVES - 1) / WAVES;
  int blocks = slots < 1024 ? slots : 1024;
  if (blocks < 1) blocks = 1;

  rbf_dqn_kernel<<<blocks, NT, 0, stream>>>(s, a_, wv1, bv1, wv2, bv2,
                                            wl1, bl1, wl2, bl2, out, Bn);
}
